// HyperConv_FD_83623013253346
// MI455X (gfx1250) — compile-verified, matrix operand order checked by reading, not run on hardware
//
#include <hip/hip_runtime.h>
#include <hip/hip_bf16.h>
#include <stddef.h>

// ---------------------------------------------------------------------------
// HyperConv fused CDNA5 (gfx1250) implementation, v2.
//   S2=32, B=32, CI=64, CO=64, L=256, K=3, A=16, H=128
// Never materialize w (402 MiB) in HBM. Per (b, 16-l tile) workgroup:
//   GEMM1 (WMMA bf16): w_chunk = h(16x128) @ W2T-chunk  -> scatter to LDS in
//                      *fragment-native* layout
//   GEMM2 (WMMA bf16): out(32s x 64o) += Xuf(32 x 32ik) @ w_lds(32ik x 64o)
//                      with both operands loaded as 2x ds_load_b128
// ---------------------------------------------------------------------------

typedef __bf16 bf16_t;
typedef __attribute__((ext_vector_type(16))) __bf16 v16bf;
typedef __attribute__((ext_vector_type(8)))  __bf16 v8bf;
typedef __attribute__((ext_vector_type(8)))  float  v8f;

#define S2n 32
#define Bn  32
#define CIn 64
#define COn 64
#define Ln  256
#define Kn  3
#define An  16
#define Hn  128
#define NW  (COn*CIn*Kn)       // 12288

// workspace layout (bytes)
#define WS_W2T   0                       // bf16 [12288][128]  = 3,145,728 B
#define WS_HBF   3145728                 // bf16 [8192][128]   = 2,097,152 B
#define WS_BIAS  5242880                 // f32  [8192][64]    = 2,097,152 B

// ---------------------------------------------------------------------------
// Kernel 0: W2 (f32, [128][12288]) -> W2T (bf16, [12288][128])
// ---------------------------------------------------------------------------
__global__ void k_prep_w2(const float* __restrict__ W2, bf16_t* __restrict__ W2T) {
    int idx = blockIdx.x * 256 + threadIdx.x;      // reads coalesced
    int h = idx / NW;
    int n = idx - h * NW;
    W2T[(size_t)n * Hn + h] = (bf16_t)W2[idx];
}

// ---------------------------------------------------------------------------
// Kernel 1: per-(b,l) hypernet rows: h = relu(LN(z@W1+b1)), bias path likewise
// ---------------------------------------------------------------------------
__global__ __launch_bounds__(128) void k_hyper(
    const float* __restrict__ z,
    const float* __restrict__ W1,  const float* __restrict__ b1,
    const float* __restrict__ g1,  const float* __restrict__ be1,
    const float* __restrict__ Wb1, const float* __restrict__ bb1,
    const float* __restrict__ gb1, const float* __restrict__ beb1,
    const float* __restrict__ Wb2, const float* __restrict__ bb2,
    bf16_t* __restrict__ h_bf, float* __restrict__ biasw) {
    __shared__ float red[128], red2[128], hbuf[128];
    const int row = blockIdx.x;            // b*L + l
    const int t   = threadIdx.x;           // 0..127

    float zr[An];
#pragma unroll
    for (int a = 0; a < An; ++a) zr[a] = z[(size_t)row * An + a];

    // ---- weight hypernet: Linear -> LN -> ReLU ----
    float acc = b1[t];
#pragma unroll
    for (int a = 0; a < An; ++a) acc = fmaf(zr[a], W1[a * Hn + t], acc);
    red[t] = acc; red2[t] = acc * acc;
    __syncthreads();
    for (int s = 64; s > 0; s >>= 1) {
        if (t < s) { red[t] += red[t + s]; red2[t] += red2[t + s]; }
        __syncthreads();
    }
    {
        float mu  = red[0] * (1.f / 128.f);
        float var = red2[0] * (1.f / 128.f) - mu * mu;
        float rs  = rsqrtf(var + 1e-5f);
        float hv  = fmaxf((acc - mu) * rs * g1[t] + be1[t], 0.f);
        h_bf[(size_t)row * Hn + t] = (bf16_t)hv;
    }
    __syncthreads();

    // ---- bias hypernet ----
    float accb = bb1[t];
#pragma unroll
    for (int a = 0; a < An; ++a) accb = fmaf(zr[a], Wb1[a * Hn + t], accb);
    red[t] = accb; red2[t] = accb * accb;
    __syncthreads();
    for (int s = 64; s > 0; s >>= 1) {
        if (t < s) { red[t] += red[t + s]; red2[t] += red2[t + s]; }
        __syncthreads();
    }
    {
        float mu  = red[0] * (1.f / 128.f);
        float var = red2[0] * (1.f / 128.f) - mu * mu;
        float rs  = rsqrtf(var + 1e-5f);
        hbuf[t]   = fmaxf((accb - mu) * rs * gb1[t] + beb1[t], 0.f);
    }
    __syncthreads();
    if (t < COn) {
        float s = bb2[t];
#pragma unroll 8
        for (int j = 0; j < Hn; ++j) s = fmaf(hbuf[j], Wb2[j * COn + t], s);
        biasw[(size_t)row * COn + t] = s;
    }
}

// ---------------------------------------------------------------------------
// Kernel 2: fused weight-generation + conv.
// grid (16 l-tiles, 32 b), 256 threads = 8 waves.
//
// Fragment lane mapping (ISA 7.12.2, 16-bit operands, wave32):
//   lane = g*16 + col ; element e in [0,16) maps to K-index
//   r(e,g) = e + 8g        for e < 8
//          = e + 8 + 8g    for e >= 8
// Inverse: g = (r>>3)&1 ; e = (r&7) + 8*(r>>4).
//
// LDS: hA   bf16[16][128]                  4 KB   (GEMM1 A rows)
//      wldF bf16[16][4 ot][32 lane][16 e]  64 KB  (GEMM2 B, fragment-native)
//      xs2  bf16[32 s][16 ll][32 r]        32 KB  (GEMM2 A, fragment-native)
// wldF is reused as f32[16][64][16] output-staging buffer after the main loop.
// ---------------------------------------------------------------------------
__global__ __launch_bounds__(256) void k_conv(
    const float*  __restrict__ x,
    const bf16_t* __restrict__ W2T,
    const float*  __restrict__ b2,
    const bf16_t* __restrict__ h_bf,
    const float*  __restrict__ biasw,
    float*        __restrict__ out) {
    __shared__ __align__(16) unsigned char smem[4096 + 65536 + 32768];
    bf16_t* hA   = (bf16_t*)smem;                    // [16][128]
    bf16_t* wldF = (bf16_t*)(smem + 4096);           // [16][4][32][16]
    bf16_t* xs2  = (bf16_t*)(smem + 4096 + 65536);   // [32][16][32]
    float*  outS = (float*)(smem + 4096);            // reuse: [16][64][16]

    const int lt = blockIdx.x;
    const int b  = blockIdx.y;
    const int l0 = lt * 16;
    const int t  = threadIdx.x;
    const int lane = t & 31;
    const int wv   = t >> 5;         // wave 0..7
    const int col  = lane & 15;
    const int grp  = lane >> 4;

    // ---- stage h rows; zero the padded k=3 entries of wldF (never rewritten)
    for (int i = t; i < 16 * Hn; i += 256)
        hA[i] = h_bf[((size_t)b * Ln + l0 + (i >> 7)) * Hn + (i & 127)];
    for (int q = t; q < 16 * 8 * 64; q += 256) {
        int l  = q >> 9;
        int il = (q >> 6) & 7;
        int o  = q & 63;
        int r  = il * 4 + 3;                               // padded tap
        int gt = (r >> 3) & 1;
        int et = (r & 7) + 8 * (r >> 4);
        wldF[((l * 4 + (o >> 4)) * 32 + gt * 16 + (o & 15)) * 16 + et] = (bf16_t)0.f;
    }
    __syncthreads();

    // ---- GEMM1 A fragments: rows = l (col), K = h.  Constant over i-blocks.
    v16bf a1[4];
#pragma unroll
    for (int kc = 0; kc < 4; ++kc) {
        union { v16bf v; v8bf h[2]; } u;
        const bf16_t* hrow = hA + col * Hn + kc * 32;
        u.h[0] = *(const v8bf*)(hrow + 8 * grp);
        u.h[1] = *(const v8bf*)(hrow + 16 + 8 * grp);
        a1[kc] = u.v;
    }

    // persistent accumulators: out[s,o] for this wave's two l's
    v8f acc[2][2][4];
    {
        v8f zf = {0.f, 0.f, 0.f, 0.f, 0.f, 0.f, 0.f, 0.f};
#pragma unroll
        for (int lp = 0; lp < 2; ++lp)
#pragma unroll
            for (int st = 0; st < 2; ++st)
#pragma unroll
                for (int ot = 0; ot < 4; ++ot) acc[lp][st][ot] = zf;
    }

    for (int ib = 0; ib < 8; ++ib) {
        const int i0 = ib * 8;

        // ======== GEMM1: w chunk = h @ W2T, scatter to wldF ================
        // N-chunk = 1536 (8 i * 192), 96 tiles of 16, 12 tiles per wave.
        const int nbase = i0 * 192;
        for (int tt = 0; tt < 12; ++tt) {
            const int tn = wv * 12 + tt;
            const int n  = nbase + tn * 16 + col;
            const bf16_t* wrow = W2T + (size_t)n * Hn;
            v8f c = {0.f, 0.f, 0.f, 0.f, 0.f, 0.f, 0.f, 0.f};
#pragma unroll
            for (int kc = 0; kc < 4; ++kc) {
                union { v16bf v; v8bf h[2]; } u;
                u.h[0] = *(const v8bf*)(wrow + kc * 32 + 8 * grp);
                u.h[1] = *(const v8bf*)(wrow + kc * 32 + 16 + 8 * grp);
                c = __builtin_amdgcn_wmma_f32_16x16x32_bf16(
                        false, a1[kc], false, u.v, (short)0, c, false, false);
            }
            // scatter: n = i*192 + o*3 + k  ->  wldF fragment slot
            const int   nl  = tn * 16 + col;               // 0..1535 in chunk
            const int   il  = nl / 192;
            const int   rem = nl - il * 192;
            const int   o   = rem / 3;
            const int   k   = rem - o * 3;
            const int   r   = il * 4 + k;
            const int   gt  = (r >> 3) & 1;
            const int   et  = (r & 7) + 8 * (r >> 4);
            const float b2v = b2[n];
            bf16_t* dst = wldF + ((o >> 4) * 32 + gt * 16 + (o & 15)) * 16 + et;
            union { v8f v; float f[8]; } uc; uc.v = c;
#pragma unroll
            for (int j = 0; j < 8; ++j) {
                int l = j + 8 * grp;                       // C row = l
                dst[l * 2048] = (bf16_t)(uc.f[j] + b2v);   // l stride = 4*32*16
            }
        }

        // ======== stage xs2: pre-unfolded im2col rows (fragment-native) =====
        // xs2[s][ll][r = i*4+k] = x[s, i0+i, l0+ll+k-1]  (k==3 -> 0)
        for (int p = t; p < 512; p += 256) {
            const int s  = p >> 4;
            const int ll = p & 15;
            const float* xrow = x + (((size_t)s * Bn + b) * CIn + i0) * Ln;
            bf16_t* dst = xs2 + (s * 16 + ll) * 32;
            const int lbase = l0 + ll - 1;
#pragma unroll
            for (int i = 0; i < 8; ++i) {
#pragma unroll
                for (int k = 0; k < 4; ++k) {
                    float v = 0.f;
                    if (k < 3) {
                        int lg = lbase + k;
                        if (lg >= 0 && lg < Ln) v = xrow[i * Ln + lg];
                    }
                    dst[i * 4 + k] = (bf16_t)v;
                }
            }
        }
        __syncthreads();

        // ======== GEMM2: out += Xuf @ wldF (all operands 2x ds_load_b128) ===
#pragma unroll
        for (int lp = 0; lp < 2; ++lp) {
            const int ll = wv * 2 + lp;
            v16bf bw[4];
#pragma unroll
            for (int ot = 0; ot < 4; ++ot) {
                union { v16bf v; v8bf h[2]; } u;
                const bf16_t* wb = wldF + ((ll * 4 + ot) * 32 + lane) * 16;
                u.h[0] = *(const v8bf*)(wb);
                u.h[1] = *(const v8bf*)(wb + 8);
                bw[ot] = u.v;
            }
            v16bf aw[2];
#pragma unroll
            for (int st = 0; st < 2; ++st) {
                union { v16bf v; v8bf h[2]; } u;
                const bf16_t* xb = xs2 + ((st * 16 + col) * 16 + ll) * 32;
                u.h[0] = *(const v8bf*)(xb + 8 * grp);
                u.h[1] = *(const v8bf*)(xb + 16 + 8 * grp);
                aw[st] = u.v;
            }
#pragma unroll
            for (int st = 0; st < 2; ++st)
#pragma unroll
                for (int ot = 0; ot < 4; ++ot)
                    acc[lp][st][ot] = __builtin_amdgcn_wmma_f32_16x16x32_bf16(
                        false, aw[st], false, bw[ot], (short)0,
                        acc[lp][st][ot], false, false);
        }
        __syncthreads();
    }

    // ======== output: transpose via LDS, add bias ===========================
    for (int st = 0; st < 2; ++st) {
#pragma unroll
        for (int lp = 0; lp < 2; ++lp) {
            const int ll = wv * 2 + lp;
#pragma unroll
            for (int ot = 0; ot < 4; ++ot) {
                const int o = ot * 16 + col;
                const float bv = biasw[((size_t)b * Ln + l0 + ll) * COn + o];
                union { v8f v; float f[8]; } uc; uc.v = acc[lp][st][ot];
#pragma unroll
                for (int j = 0; j < 8; ++j) {
                    int sl = j + 8 * grp;
                    outS[(sl * 64 + o) * 16 + ll] = uc.f[j] + bv;
                }
            }
        }
        __syncthreads();
        for (int rr = t; rr < 1024; rr += 256) {
            int sl = rr >> 6;
            int o  = rr & 63;
            int s  = st * 16 + sl;
            float* op = out + (((size_t)s * Bn + b) * COn + o) * Ln + l0;
            const float* ip = outS + (sl * 64 + o) * 16;
            *(float4*)(op)      = *(const float4*)(ip);
            *(float4*)(op + 4)  = *(const float4*)(ip + 4);
            *(float4*)(op + 8)  = *(const float4*)(ip + 8);
            *(float4*)(op + 12) = *(const float4*)(ip + 12);
        }
        __syncthreads();
    }
}

// ---------------------------------------------------------------------------
extern "C" void kernel_launch(void* const* d_in, const int* in_sizes, int n_in,
                              void* d_out, int out_size, void* d_ws, size_t ws_size,
                              hipStream_t stream) {
    const float* x    = (const float*)d_in[0];
    const float* z    = (const float*)d_in[1];
    const float* W1   = (const float*)d_in[2];
    const float* b1   = (const float*)d_in[3];
    const float* g1   = (const float*)d_in[4];
    const float* be1  = (const float*)d_in[5];
    const float* W2   = (const float*)d_in[6];
    const float* b2   = (const float*)d_in[7];
    const float* Wb1  = (const float*)d_in[8];
    const float* bb1  = (const float*)d_in[9];
    const float* gb1  = (const float*)d_in[10];
    const float* beb1 = (const float*)d_in[11];
    const float* Wb2  = (const float*)d_in[12];
    const float* bb2  = (const float*)d_in[13];

    bf16_t* W2T   = (bf16_t*)((char*)d_ws + WS_W2T);
    bf16_t* h_bf  = (bf16_t*)((char*)d_ws + WS_HBF);
    float*  biasw = (float*)((char*)d_ws + WS_BIAS);

    k_prep_w2<<<6144, 256, 0, stream>>>(W2, W2T);
    k_hyper<<<Bn * Ln, 128, 0, stream>>>(z, W1, b1, g1, be1,
                                         Wb1, bb1, gb1, beb1, Wb2, bb2,
                                         h_bf, biasw);
    k_conv<<<dim3(Ln / 16, Bn), 256, 0, stream>>>(x, W2T, b2, h_bf, biasw,
                                                  (float*)d_out);
}
